// GPT2PseudoAttention_45938970198191
// MI455X (gfx1250) — matrix operand order repaired and checked by reading, not compiled, not run on hardware
//
#include <hip/hip_runtime.h>
#include <hip/hip_bf16.h>
#include <stdint.h>

// GPT2 pseudo-attention, MI455X (gfx1250), wave32, bf16 WMMA + TDM pipeline.
#define EMB     1024
#define NHEADS  16
#define HD      64
#define BATCH   4
#define SEQ     1023
#define KLEN    1024                 // SEQ + 1 image token
#define MROWS   (BATCH * SEQ)        // 4092
#define MASK_VALF (-65504.0f)        // float16 min, per reference

typedef __attribute__((ext_vector_type(16))) __bf16 v16bf;
typedef __attribute__((ext_vector_type(8)))  float  v8f;
typedef __attribute__((ext_vector_type(4)))  uint32_t u32x4;
typedef __attribute__((ext_vector_type(8)))  uint32_t u32x8;

__device__ __forceinline__ unsigned short f2bf(float f) {
  union { float f; uint32_t u; } c; c.f = f;
  uint32_t r = c.u + 0x7FFFu + ((c.u >> 16) & 1u);   // round-to-nearest-even
  return (unsigned short)(r >> 16);
}

// ---------------------------------------------------------------------------
// Tensor Data Mover: 2D tile load Global->LDS (ISA 08_async_tensor D# layout).
// data_size = 4B (rows expressed in DWORDs). pad_enable inserts 1 DWORD of
// LDS padding every (2<<pad_code) DWORDs  ->  reproduces the +1-DWORD row
// padding used for bank-conflict-free fragment reads.
// Issue from one wave; EXEC is ignored by TDM. Tracked by TENSORcnt.
// ---------------------------------------------------------------------------
__device__ __forceinline__ void tdm_load_2d(uint32_t lds_off, const void* gptr,
                                            uint32_t row_dwords,   // tile/tensor dim0
                                            uint32_t tile_rows,    // tile dim1
                                            uint32_t tensor_rows,  // OOB rows read 0
                                            uint32_t stride_dwords,
                                            uint32_t pad_code) {
  const uint64_t ga = (uint64_t)(uintptr_t)gptr;
  u32x4 g0;
  g0[0] = 1u;                                              // count=1 (valid, user)
  g0[1] = lds_off;                                         // lds_addr (bytes)
  g0[2] = (uint32_t)ga;                                    // global_addr[31:0]
  g0[3] = ((uint32_t)(ga >> 32) & 0x01FFFFFFu) | (2u << 30); // [56:32] | type=2
  u32x8 g1;
  g1[0] = (2u << 16) | (1u << 20) | (pad_code << 22);      // 4B, pad_en, interval; pad=1 DW
  g1[1] = (row_dwords & 0xFFFFu) << 16;                    // tensor_dim0 lo
  g1[2] = ((row_dwords >> 16) & 0xFFFFu) | ((tensor_rows & 0xFFFFu) << 16);
  g1[3] = ((tensor_rows >> 16) & 0xFFFFu) | ((row_dwords & 0xFFFFu) << 16); // tile_dim0
  g1[4] = tile_rows & 0xFFFFu;                             // tile_dim1 (dim2=0)
  g1[5] = stride_dwords;                                   // tensor_dim0_stride lo
  g1[6] = 0u;
  g1[7] = 0u;
  const u32x4 gz = {0u, 0u, 0u, 0u};                       // groups 2/3 unused (2D)
  asm volatile("tensor_load_to_lds %0, %1, %2, %3"
               :: "s"(g0), "s"(g1), "s"(gz), "s"(gz)
               : "memory");
}

__device__ __forceinline__ uint32_t lds_off32(const void* p) {
  return (uint32_t)(uintptr_t)p;   // LDS aperture: addr[31:0] is the LDS offset
}

// A-fragment (16x32 bf16) per ISA 7.12.2: lanes 0-15 row m=lane, K {0-7,16-23};
// lanes 16-31 row m=lane-16, K {8-15,24-31}; pairs packed per VGPR.
__device__ __forceinline__ v16bf load_a_frag(const unsigned short* base, int ld,
                                             int mrow, int kcol, int lane) {
  union { v16bf v; uint32_t u[8]; } r;
  const unsigned short* row = base + (size_t)(mrow + (lane & 15)) * ld + kcol;
  const int kh = (lane & 16) ? 8 : 0;
#pragma unroll
  for (int j = 0; j < 4; ++j) r.u[j]     = *(const uint32_t*)(row + kh + 2 * j);
#pragma unroll
  for (int j = 0; j < 4; ++j) r.u[4 + j] = *(const uint32_t*)(row + 16 + kh + 2 * j);
  return r.v;
}

// B-fragment (32x16 bf16): lanes 0-15 col n=lane, K {0-7,16-23}; lanes 16-31 col
// n=lane-16, K {8-15,24-31}. LDS layout [n][k] (k = contraction dim, contiguous).
__device__ __forceinline__ v16bf load_b_frag(const unsigned short* base, int ld,
                                             int ncol, int kcol, int lane) {
  union { v16bf v; uint32_t u[8]; } r;
  const int kh = (lane & 16) ? 8 : 0;
  const unsigned short* row = base + (size_t)(ncol + (lane & 15)) * ld + kcol;
#pragma unroll
  for (int j = 0; j < 4; ++j) r.u[j]     = *(const uint32_t*)(row + kh + 2 * j);
#pragma unroll
  for (int j = 0; j < 4; ++j) r.u[4 + j] = *(const uint32_t*)(row + 16 + kh + 2 * j);
  return r.v;
}

__device__ __forceinline__ v8f wmma_bf16(v16bf a, v16bf b, v8f c) {
  return __builtin_amdgcn_wmma_f32_16x16x32_bf16(false, a, false, b,
                                                 (short)0, c, false, false);
}

// ---------------------------------------------------------------------------
// Generic 64x64-tile GEMM: C = A(Mx1024) * W(1024xN) + bias, bf16 WMMA.
// EPI==0: A is f32 (word hidden states); scatter q/k/v bf16 with split-heads
//         and image-token offset (+1 in key dim). Manual A staging (converts).
// EPI==1: A is bf16 (attn scratch), staged via TDM; store f32 to d_out.
// ---------------------------------------------------------------------------
template <int EPI>
__global__ __launch_bounds__(128) void gemm_kernel(
    const void* __restrict__ Aptr, const float* __restrict__ W,
    const float* __restrict__ bias, float* __restrict__ outF,
    unsigned short* __restrict__ qbf, unsigned short* __restrict__ kbf,
    unsigned short* __restrict__ vbf, int Ncols) {
  __shared__ unsigned short aS[64][34];   // [m][k], 16 DWORDs + 1 pad per row
  __shared__ unsigned short bS[64][34];   // [n][k] (transposed on fill)

  const int tid  = threadIdx.x;
  const int lane = tid & 31;
  const int wave = tid >> 5;
  const int colbase = blockIdx.x * 64;
  const int rowbase = blockIdx.y * 64;

  v8f acc[4] = {};

  for (int kt = 0; kt < EMB / 32; ++kt) {
    // ---- A tile: 64 rows x 32 k ----
    if (EPI == 0) {
      const float* A = (const float*)Aptr;
#pragma unroll
      for (int i = 0; i < 4; ++i) {
        int slot = tid + i * 128;              // 512 slots of float4
        int row = slot >> 3;
        int c4  = (slot & 7) << 2;
        int gr  = rowbase + row; if (gr > MROWS - 1) gr = MROWS - 1;
        const float4 f = *(const float4*)(A + (size_t)gr * EMB + kt * 32 + c4);
        *(uint32_t*)&aS[row][c4]     = ((uint32_t)f2bf(f.y) << 16) | f2bf(f.x);
        *(uint32_t*)&aS[row][c4 + 2] = ((uint32_t)f2bf(f.w) << 16) | f2bf(f.z);
      }
    } else if (wave == 0) {
      // bf16 rows of 32 elems (16 DWORDs), global row stride 512 DWORDs.
      const unsigned short* A = (const unsigned short*)Aptr;
      tdm_load_2d(lds_off32(&aS[0][0]),
                  A + (size_t)rowbase * EMB + kt * 32,
                  /*row_dwords=*/16, /*tile_rows=*/64,
                  /*tensor_rows=*/(uint32_t)(MROWS - rowbase),
                  /*stride_dwords=*/EMB / 2, /*pad_code=*/3);   // 2<<3 = 16 DW
    }
    // ---- B tile: 32 k x 64 n, f32 -> bf16, stored transposed [n][k] ----
#pragma unroll
    for (int i = 0; i < 4; ++i) {
      int slot = tid + i * 128;                // 512 slots of float4
      int k  = slot >> 4;
      int n4 = (slot & 15) << 2;
      const float4 f = *(const float4*)(W + (size_t)(kt * 32 + k) * Ncols + colbase + n4);
      bS[n4 + 0][k] = f2bf(f.x);
      bS[n4 + 1][k] = f2bf(f.y);
      bS[n4 + 2][k] = f2bf(f.z);
      bS[n4 + 3][k] = f2bf(f.w);
    }
    if (EPI == 1 && wave == 0) __builtin_amdgcn_s_wait_tensorcnt(0);
    __syncthreads();

    v16bf a = load_a_frag(&aS[0][0], 34, wave * 16, 0, lane);
#pragma unroll
    for (int t = 0; t < 4; ++t)
      acc[t] = wmma_bf16(a, load_b_frag(&bS[0][0], 34, t * 16, 0, lane), acc[t]);
    __syncthreads();
  }

  // ---- epilogue; C layout: VGPR i -> row i (lanes 0-15) / i+8 (lanes 16-31)
  const int rsel = (lane & 16) ? 8 : 0;
#pragma unroll
  for (int t = 0; t < 4; ++t) {
#pragma unroll
    for (int i = 0; i < 8; ++i) {
      const int r = rowbase + wave * 16 + i + rsel;
      if (r >= MROWS) continue;
      const int n = colbase + t * 16 + (lane & 15);
      const float val = acc[t][i] + bias[n];
      if (EPI == 0) {
        const int b = r / SEQ, s = r % SEQ;
        const int which = n >> 10;
        const int e = n & 1023;
        const int h = e >> 6, d = e & 63;
        const unsigned short bv = f2bf(val);
        if (which == 0)
          qbf[(((size_t)(b * NHEADS + h)) * SEQ + s) * HD + d] = bv;
        else if (which == 1)
          kbf[(((size_t)(b * NHEADS + h)) * KLEN + (s + 1)) * HD + d] = bv;
        else
          vbf[(((size_t)(b * NHEADS + h)) * KLEN + (s + 1)) * HD + d] = bv;
      } else {
        outF[(size_t)r * EMB + n] = val;
      }
    }
  }
}

// ---------------------------------------------------------------------------
// Image token K/V: k_img[b,e] = img[b,:] . uk_w[e,:] + uk_b[e]  (and v)
// ---------------------------------------------------------------------------
__global__ __launch_bounds__(256) void image_kv_kernel(
    const float* __restrict__ img, const float* __restrict__ ukw,
    const float* __restrict__ ukb, const float* __restrict__ uvw,
    const float* __restrict__ uvb, unsigned short* __restrict__ kbf,
    unsigned short* __restrict__ vbf) {
  const int idx = blockIdx.x * blockDim.x + threadIdx.x;   // 0..4095
  const int b = idx >> 10;
  const int c = idx & 1023;
  const float* x  = img + (size_t)b * EMB;
  const float* wk = ukw + (size_t)c * EMB;
  const float* wv = uvw + (size_t)c * EMB;
  float ak = ukb[c], av = uvb[c];
  for (int j = 0; j < EMB; ++j) {
    const float xv = x[j];
    ak += xv * wk[j];
    av += xv * wv[j];
  }
  const int h = c >> 6, d = c & 63;
  kbf[(((size_t)(b * NHEADS + h)) * KLEN) * HD + d] = f2bf(ak);
  vbf[(((size_t)(b * NHEADS + h)) * KLEN) * HD + d] = f2bf(av);
}

// ---------------------------------------------------------------------------
// Flash-style causal attention per (b,h): 64 q rows / block, 32 keys / step.
// Q and K tiles staged by TDM (with hardware LDS row padding); V staged
// manually (needs transpose for the P.V B-fragment layout).
// ---------------------------------------------------------------------------
__global__ __launch_bounds__(128) void attn_kernel(
    const unsigned short* __restrict__ qbf, const unsigned short* __restrict__ kbf,
    const unsigned short* __restrict__ vbf, const float* __restrict__ amask,
    unsigned short* __restrict__ attnbf) {
  __shared__ unsigned short qS[64][66];      // [q row][d]   32 DW + 1 pad
  __shared__ unsigned short kS[32][66];      // [kpos][d]    (B-frag for q.k^T)
  __shared__ unsigned short vS[64][34];      // [d][kpos]    (B-frag for P.V)
  __shared__ unsigned short pS[4][16][34];   // per-wave probs bounce [m][kpos]
  __shared__ float aMs[32];

  const int tid  = threadIdx.x;
  const int lane = tid & 31;
  const int wave = tid >> 5;
  const int bh = blockIdx.x;
  const int b  = bh / NHEADS;
  const int h  = bh % NHEADS;
  const int qb = blockIdx.y * 64;

  const size_t qbase  = (size_t)bh * SEQ  * HD;
  const size_t kvbase = (size_t)bh * KLEN * HD;

  // Q tile 64x64 via TDM; OOB rows (last block) return zeros.
  if (wave == 0) {
    tdm_load_2d(lds_off32(&qS[0][0]), qbf + qbase + (size_t)qb * HD,
                /*row_dwords=*/32, /*tile_rows=*/64,
                /*tensor_rows=*/(uint32_t)(SEQ - qb),
                /*stride_dwords=*/32, /*pad_code=*/4);          // 2<<4 = 32 DW
  }

  float rowm[8], rowl[8];
  v8f accv[4] = {};
#pragma unroll
  for (int i = 0; i < 8; ++i) { rowm[i] = -1e30f; rowl[i] = 0.0f; }

  int kmax = qb + 65;                          // causal: keys up to qi+1
  if (kmax > KLEN) kmax = KLEN;
  const int nkb = (kmax + 31) >> 5;

  const int nl   = lane & 15;
  const int rsel = (lane & 16) ? 8 : 0;

  if (wave == 0) __builtin_amdgcn_s_wait_tensorcnt(0);
  __syncthreads();

  for (int kt = 0; kt < nkb; ++kt) {
    const int kb = kt * 32;
    // K tile 32x64 via TDM (contiguous rows, stride 32 DW).
    if (wave == 0) {
      tdm_load_2d(lds_off32(&kS[0][0]), kbf + kvbase + (size_t)kb * HD,
                  /*row_dwords=*/32, /*tile_rows=*/32, /*tensor_rows=*/32,
                  /*stride_dwords=*/32, /*pad_code=*/4);
    }
    // V tile 32x64 -> transposed [d][kpos] (manual; TDM cannot transpose).
#pragma unroll
    for (int i = 0; i < 4; ++i) {
      int slot = tid + i * 128;
      int kk = slot >> 4;
      int c4 = (slot & 15) << 2;
      const uint2 u = *(const uint2*)(vbf + kvbase + (size_t)(kb + kk) * HD + c4);
      vS[c4 + 0][kk] = (unsigned short)(u.x & 0xFFFFu);
      vS[c4 + 1][kk] = (unsigned short)(u.x >> 16);
      vS[c4 + 2][kk] = (unsigned short)(u.y & 0xFFFFu);
      vS[c4 + 3][kk] = (unsigned short)(u.y >> 16);
    }
    if (tid < 32) aMs[tid] = amask[(size_t)b * KLEN + kb + tid];
    if (wave == 0) __builtin_amdgcn_s_wait_tensorcnt(0);
    __syncthreads();

    // scores: two 16x16 key tiles, d=64 contraction -> 2 WMMA K-steps each
    v16bf a0 = load_a_frag(&qS[0][0], 66, wave * 16, 0,  lane);
    v16bf a1 = load_a_frag(&qS[0][0], 66, wave * 16, 32, lane);
    v8f s0 = {}, s1 = {};
    s0 = wmma_bf16(a0, load_b_frag(&kS[0][0], 66, 0,  0,  lane), s0);
    s0 = wmma_bf16(a1, load_b_frag(&kS[0][0], 66, 0,  32, lane), s0);
    s1 = wmma_bf16(a0, load_b_frag(&kS[0][0], 66, 16, 0,  lane), s1);
    s1 = wmma_bf16(a1, load_b_frag(&kS[0][0], 66, 16, 32, lane), s1);

    const float am0 = aMs[nl];
    const float am1 = aMs[nl + 16];

#pragma unroll
    for (int i = 0; i < 8; ++i) {
      const int qi = qb + wave * 16 + i + rsel;
      float x0 = ((kb + nl)      <= qi + 1 ? s0[i] * 0.125f : MASK_VALF) + am0;
      float x1 = ((kb + 16 + nl) <= qi + 1 ? s1[i] * 0.125f : MASK_VALF) + am1;
      float tm = fmaxf(x0, x1);
#pragma unroll
      for (int off = 8; off; off >>= 1) tm = fmaxf(tm, __shfl_xor(tm, off, 32));
      const float newm  = fmaxf(rowm[i], tm);
      const float alpha = __expf(rowm[i] - newm);
      const float p0 = __expf(x0 - newm);
      const float p1 = __expf(x1 - newm);
      float ps = p0 + p1;
#pragma unroll
      for (int off = 8; off; off >>= 1) ps += __shfl_xor(ps, off, 32);
      rowl[i] = rowl[i] * alpha + ps;
      rowm[i] = newm;
#pragma unroll
      for (int t = 0; t < 4; ++t) accv[t][i] *= alpha;
      pS[wave][i + rsel][nl]      = f2bf(p0);
      pS[wave][i + rsel][nl + 16] = f2bf(p1);
    }
    __syncthreads();   // C-layout -> A-layout bounce through LDS

    v16bf pa = load_a_frag(&pS[wave][0][0], 34, 0, 0, lane);
#pragma unroll
    for (int t = 0; t < 4; ++t)
      accv[t] = wmma_bf16(pa, load_b_frag(&vS[0][0], 34, t * 16, 0, lane), accv[t]);
    __syncthreads();
  }

  // normalize & store attn in [b][s][h*64+d] (bf16, GEMM-A layout for proj)
#pragma unroll
  for (int t = 0; t < 4; ++t) {
#pragma unroll
    for (int i = 0; i < 8; ++i) {
      const int qi = qb + wave * 16 + i + rsel;
      if (qi >= SEQ) continue;
      const float val = accv[t][i] / rowl[i];
      attnbf[((size_t)(b * SEQ + qi)) * EMB + h * HD + t * 16 + nl] = f2bf(val);
    }
  }
}

// ---------------------------------------------------------------------------
extern "C" void kernel_launch(void* const* d_in, const int* in_sizes, int n_in,
                              void* d_out, int out_size, void* d_ws, size_t ws_size,
                              hipStream_t stream) {
  (void)in_sizes; (void)n_in; (void)out_size; (void)ws_size;
  const float* X     = (const float*)d_in[0];   // word_hidden_states [4,1023,1024]
  const float* img   = (const float*)d_in[1];   // image_hidden_states [4,1024]
  const float* amask = (const float*)d_in[2];   // attention_mask [4,1,1,1024]
  const float* caw   = (const float*)d_in[3];   // c_attn_w [1024,3072]
  const float* cab   = (const float*)d_in[4];   // c_attn_b [3072]
  const float* cpw   = (const float*)d_in[5];   // c_proj_w [1024,1024]
  const float* cpb   = (const float*)d_in[6];   // c_proj_b [1024]
  const float* ukw   = (const float*)d_in[7];
  const float* ukb   = (const float*)d_in[8];
  const float* uvw   = (const float*)d_in[9];
  const float* uvb   = (const float*)d_in[10];
  float* out = (float*)d_out;

  // bf16 scratch in d_ws (~33.5 MB total; resident in 192 MB L2)
  unsigned short* qbf    = (unsigned short*)d_ws;                       // [b,h,s,d]
  unsigned short* kbf    = qbf + (size_t)BATCH * NHEADS * SEQ  * HD;    // [b,h,k,d]
  unsigned short* vbf    = kbf + (size_t)BATCH * NHEADS * KLEN * HD;    // [b,h,k,d]
  unsigned short* attnbf = vbf + (size_t)BATCH * NHEADS * KLEN * HD;    // [b*s, e]

  dim3 g1(3 * EMB / 64, (MROWS + 63) / 64);
  gemm_kernel<0><<<g1, 128, 0, stream>>>(X, caw, cab, nullptr, qbf, kbf, vbf, 3 * EMB);

  image_kv_kernel<<<(BATCH * EMB) / 256, 256, 0, stream>>>(img, ukw, ukb, uvw, uvb,
                                                           kbf, vbf);

  dim3 g2(BATCH * NHEADS, (SEQ + 63) / 64);
  attn_kernel<<<g2, 128, 0, stream>>>(qbf, kbf, vbf, amask, attnbf);

  dim3 g3(EMB / 64, (MROWS + 63) / 64);
  gemm_kernel<1><<<g3, 128, 0, stream>>>(attnbf, cpw, cpb, out,
                                         nullptr, nullptr, nullptr, EMB);
}
